// RMultiHeadGraphAttention_52716428591535
// MI455X (gfx1250) — matrix operand order repaired and checked by reading, not compiled, not run on hardware
//
#include <hip/hip_runtime.h>
#include <hip/hip_bf16.h>

#define NN 50000
#define NE 800000
#define FF 128
#define NTILE (NN / 16)   // 3125 exactly

typedef float     v2f  __attribute__((ext_vector_type(2)));
typedef float     v8f  __attribute__((ext_vector_type(8)));
typedef _Float16  v16h __attribute__((ext_vector_type(16)));

// ---------------------------------------------------------------------------
// 1) Fold heads: B[col][f] = w_eff[i][f] * a[i][k][f], col = 3*i + k, padded
//    to 16 columns with zeros (head 3 reuses w[2]).
// ---------------------------------------------------------------------------
__global__ void build_b_kernel(const float* __restrict__ w,
                               const float* __restrict__ a,
                               float* __restrict__ B) {
    int t = blockIdx.x * blockDim.x + threadIdx.x;
    if (t >= 16 * FF) return;
    int col = t / FF, f = t % FF;
    float v = 0.0f;
    if (col < 12) {
        int i = col / 3, k = col % 3;
        int wi = (i < 3) ? i : 2;
        v = w[wi * FF + f] * a[(i * 3 + k) * FF + f];
    }
    B[col * FF + f] = v;
}

// ---------------------------------------------------------------------------
// 2) Score GEMM  S[N,12] = x[N,128] @ B^T[128,16(12)]  via WMMA.
//    One wave per 16-node tile. 3125 tiles (N divisible by 16).
// ---------------------------------------------------------------------------
#if __has_builtin(__builtin_amdgcn_wmma_f32_16x16x4_f32)
__global__ void score_wmma_kernel(const float* __restrict__ x,
                                  const float* __restrict__ B,
                                  float* __restrict__ S) {
    int wave = (blockIdx.x * blockDim.x + threadIdx.x) >> 5;
    int lane = threadIdx.x & 31;
    if (wave >= NTILE) return;
    int base  = wave * 16;
    int m     = lane & 15;            // A row / B col handled by this lane
    int khalf = (lane >> 4) << 1;     // 0 or 2  (f32 16x4 A layout)
    const float* xrow = x + (size_t)(base + m) * FF;
    const float* bcol = B + (size_t)m * FF;
    v8f c = {};
    #pragma unroll 4
    for (int k0 = 0; k0 < FF; k0 += 4) {
        v2f af, bf;
        af.x = xrow[k0 + khalf];
        af.y = xrow[k0 + khalf + 1];
        bf.x = bcol[k0 + khalf];
        bf.y = bcol[k0 + khalf + 1];
        c = __builtin_amdgcn_wmma_f32_16x16x4_f32(false, af, false, bf,
                                                  (short)0, c, false, false);
    }
    int col = lane & 15;
    if (col < 12) {
        int rowoff = (lane >> 4) * 8;   // C layout: lanes16-31 hold M+8
        #pragma unroll
        for (int r = 0; r < 8; ++r)
            S[(size_t)(base + rowoff + r) * 12 + col] = c[r];
    }
}
#else
// Fallback: f16 WMMA 16x16x32 (codegen-confirmed builtin); x~N(0,1) so f16
// conversion of A/B keeps ~1e-3 relative accuracy on the scores.
__global__ void score_wmma_kernel(const float* __restrict__ x,
                                  const float* __restrict__ B,
                                  float* __restrict__ S) {
    int wave = (blockIdx.x * blockDim.x + threadIdx.x) >> 5;
    int lane = threadIdx.x & 31;
    if (wave >= NTILE) return;
    int base = wave * 16;
    int m    = lane & 15;
    int khiA = (lane >> 4) * 8;    // 16-bit A layout: lanes16-31 start at K+8
    int khiB = (lane >> 4) * 16;   // k order used here for A and B: lanes16-31 hold K+16
    const float* xrow = x + (size_t)(base + m) * FF;
    const float* bcol = B + (size_t)m * FF;
    v8f c = {};
    for (int k0 = 0; k0 < FF; k0 += 32) {
        v16h af, bf;
        #pragma unroll
        for (int v = 0; v < 8; ++v) {
            int ka = k0 + khiB + 2 * v;
            af[2 * v]     = (_Float16)xrow[ka];
            af[2 * v + 1] = (_Float16)xrow[ka + 1];
            int kb = k0 + khiB + 2 * v;
            bf[2 * v]     = (_Float16)bcol[kb];
            bf[2 * v + 1] = (_Float16)bcol[kb + 1];
        }
        c = __builtin_amdgcn_wmma_f32_16x16x32_f16(false, af, false, bf,
                                                   (short)0, c, false, false);
    }
    int col = lane & 15;
    if (col < 12) {
        int rowoff = (lane >> 4) * 8;
        #pragma unroll
        for (int r = 0; r < 8; ++r)
            S[(size_t)(base + rowoff + r) * 12 + col] = c[r];
    }
}
#endif

// ---------------------------------------------------------------------------
// 3) CSR build: zero, count by src, exclusive scan, scatter permutation.
// ---------------------------------------------------------------------------
__global__ void zero_u32_kernel(unsigned* __restrict__ p, int n) {
    int i = blockIdx.x * blockDim.x + threadIdx.x;
    if (i < n) p[i] = 0u;
}

__global__ void count_kernel(const long long* __restrict__ A,
                             unsigned* __restrict__ cnt) {
    int e = blockIdx.x * blockDim.x + threadIdx.x;
    if (e >= NE) return;
    atomicAdd(&cnt[(int)A[e]], 1u);
}

__global__ void scan_kernel(const unsigned* __restrict__ deg,
                            unsigned* __restrict__ rowptr, int n) {
    __shared__ unsigned tmp[1024];
    __shared__ unsigned running_s;
    int tid = threadIdx.x;
    if (tid == 0) running_s = 0u;
    __syncthreads();
    for (int basei = 0; basei < n; basei += 1024) {
        int i = basei + tid;
        unsigned v = (i < n) ? deg[i] : 0u;
        tmp[tid] = v;
        __syncthreads();
        for (int off = 1; off < 1024; off <<= 1) {
            unsigned add = (tid >= off) ? tmp[tid - off] : 0u;
            __syncthreads();
            tmp[tid] += add;
            __syncthreads();
        }
        unsigned incl = tmp[tid];
        unsigned run  = running_s;
        if (i < n) rowptr[i] = run + incl - v;   // exclusive
        __syncthreads();
        if (tid == 1023) running_s = run + tmp[1023];
        __syncthreads();
    }
    if (tid == 0) rowptr[n] = running_s;
}

__global__ void scatter_kernel(const long long* __restrict__ A,
                               const unsigned* __restrict__ rowptr,
                               unsigned* __restrict__ cnt,
                               unsigned* __restrict__ perm) {
    int e = blockIdx.x * blockDim.x + threadIdx.x;
    if (e >= NE) return;
    int s = (int)A[e];
    unsigned pos = rowptr[s] + atomicAdd(&cnt[s], 1u);
    perm[pos] = (unsigned)e;
}

// ---------------------------------------------------------------------------
// 4) Per-edge attention weights for all 4 heads packed as float4.
// ---------------------------------------------------------------------------
__device__ __forceinline__ float neg_lrelu_exp(float eh) {
    float lr = (eh >= 0.0f) ? eh : 0.2f * eh;
    return expf(-lr);
}

__global__ void edge_score_kernel(const long long* __restrict__ A,
                                  const long long* __restrict__ inputr,
                                  const float* __restrict__ S,
                                  float4* __restrict__ ee) {
    int e = blockIdx.x * blockDim.x + threadIdx.x;
    if (e >= NE) return;
    int s = (int)A[e];
    int d = (int)A[(size_t)2 * NE + e];
    int r = (int)inputr[(size_t)NE + e];   // reference uses inputr[1]
    const float* Ss = S + (size_t)s * 12;
    const float* Sd = S + (size_t)d * 12;
    const float* Sr = S + (size_t)r * 12;
    float4 o;
    o.x = neg_lrelu_exp(Ss[0] + Sd[1]  + Sr[2]);
    o.y = neg_lrelu_exp(Ss[3] + Sd[4]  + Sr[5]);
    o.z = neg_lrelu_exp(Ss[6] + Sd[7]  + Sr[8]);
    o.w = neg_lrelu_exp(Ss[9] + Sd[10] + Sr[11]);
    ee[e] = o;
}

// ---------------------------------------------------------------------------
// 5) Gather-aggregate: one wave per node, lane l owns features 4l..4l+3,
//    all 4 heads accumulated in registers; rowsum uniform per wave.
//    out[i][n][f] = w_eff[i][f] * acc_i[f] / rowsum_i     (no float atomics)
// ---------------------------------------------------------------------------
__global__ void aggregate_kernel(const float* __restrict__ x,
                                 const float* __restrict__ w,
                                 const long long* __restrict__ A,
                                 const unsigned* __restrict__ rowptr,
                                 const unsigned* __restrict__ perm,
                                 const float4* __restrict__ ee,
                                 float* __restrict__ out) {
    int wave = (blockIdx.x * blockDim.x + threadIdx.x) >> 5;
    int lane = threadIdx.x & 31;
    if (wave >= NN) return;
    int n  = wave;
    int f0 = lane * 4;
    const long long* dstArr = A + (size_t)2 * NE;

    float4 acc[4];
    float  rs[4];
    #pragma unroll
    for (int i = 0; i < 4; ++i) { acc[i] = make_float4(0.f, 0.f, 0.f, 0.f); rs[i] = 0.f; }

    unsigned jb = rowptr[n], je = rowptr[n + 1];
    for (unsigned j = jb; j < je; ++j) {
        unsigned e = perm[j];                                   // uniform
        int d = (int)dstArr[e];                                 // uniform
        float4 we = ee[e];                                      // uniform 16B
        float  wh[4] = { we.x, we.y, we.z, we.w };
        float4 xv = *(const float4*)(x + (size_t)d * FF + f0);  // per-lane 16B
        #pragma unroll
        for (int i = 0; i < 4; ++i) {
            acc[i].x += wh[i] * xv.x;
            acc[i].y += wh[i] * xv.y;
            acc[i].z += wh[i] * xv.z;
            acc[i].w += wh[i] * xv.w;
            rs[i]    += wh[i];
        }
    }
    #pragma unroll
    for (int i = 0; i < 4; ++i) {
        int wi = (i < 3) ? i : 2;
        float4 wv = *(const float4*)(w + (size_t)wi * FF + f0);
        float inv = 1.0f / rs[i];
        float4 o;
        o.x = wv.x * acc[i].x * inv;
        o.y = wv.y * acc[i].y * inv;
        o.z = wv.z * acc[i].z * inv;
        o.w = wv.w * acc[i].w * inv;
        *(float4*)(out + ((size_t)i * NN + n) * FF + f0) = o;
    }
}

// ---------------------------------------------------------------------------
// Launch
// ---------------------------------------------------------------------------
extern "C" void kernel_launch(void* const* d_in, const int* in_sizes, int n_in,
                              void* d_out, int out_size, void* d_ws, size_t ws_size,
                              hipStream_t stream) {
    const float*     x   = (const float*)d_in[0];       // [N,128]
    const float*     w   = (const float*)d_in[1];       // [3,1,128]
    const float*     a   = (const float*)d_in[2];       // [4,3,128,1]
    const long long* A   = (const long long*)d_in[3];   // [3,E] int64
    const long long* ir  = (const long long*)d_in[4];   // [2,E] int64
    float*           out = (float*)d_out;               // [4,N,128]

    // Workspace layout (all 16B aligned): ~18.8 MB total
    char* ws = (char*)d_ws;
    float*    B      = (float*)   (ws + 0);          //  8,192 B
    float*    S      = (float*)   (ws + 8192);       //  2,400,000 B
    unsigned* rowptr = (unsigned*)(ws + 2408192);    //  200,004 B
    unsigned* cnt    = (unsigned*)(ws + 2608208);    //  200,000 B
    unsigned* perm   = (unsigned*)(ws + 2808208);    //  3,200,000 B
    float4*   ee     = (float4*)  (ws + 6008208);    // 12,800,000 B

    build_b_kernel<<<(16 * FF + 255) / 256, 256, 0, stream>>>(w, a, B);

    // one wave (32 lanes) per 16-row tile, 4 waves / block
    score_wmma_kernel<<<(NTILE + 3) / 4, 128, 0, stream>>>(x, B, S);

    zero_u32_kernel<<<(NN + 255) / 256, 256, 0, stream>>>(cnt, NN);
    count_kernel<<<(NE + 255) / 256, 256, 0, stream>>>(A, cnt);
    scan_kernel<<<1, 1024, 0, stream>>>(cnt, rowptr, NN);
    zero_u32_kernel<<<(NN + 255) / 256, 256, 0, stream>>>(cnt, NN);
    scatter_kernel<<<(NE + 255) / 256, 256, 0, stream>>>(A, rowptr, cnt, perm);

    edge_score_kernel<<<(NE + 255) / 256, 256, 0, stream>>>(A, ir, S, ee);

    // one wave per node, 8 waves / block
    aggregate_kernel<<<(NN + 7) / 8, 256, 0, stream>>>(x, w, A, rowptr, perm, ee, out);
}
